// RoPEBlock_73976516707036
// MI455X (gfx1250) — matrix operand order repaired and checked by reading, not compiled, not run on hardware
//
#include <hip/hip_runtime.h>
#include <hip/hip_bf16.h>

// MI455X / gfx1250, wave32. Matmuls via v_wmma_f32_16x16x32_bf16,
// q-strip staging via global_load_async_to_lds_b128 (ASYNCcnt).
#define EMBED 512
#define TLEN  512
#define BATCH 16

typedef __attribute__((ext_vector_type(16))) __bf16       v16bf;
typedef __attribute__((ext_vector_type(8)))  float        v8f;
typedef __attribute__((ext_vector_type(8)))  unsigned int v8u;

__device__ __forceinline__ unsigned short f2bf(float f) {
    unsigned u = __float_as_uint(f);
    u += 0x7FFFu + ((u >> 16) & 1u);   // round-to-nearest-even
    return (unsigned short)(u >> 16);
}

__device__ __forceinline__ v8f bf16_wmma(const v8u& au, const v8u& bu, const v8f& c) {
    v16bf a = __builtin_bit_cast(v16bf, au);
    v16bf b = __builtin_bit_cast(v16bf, bu);
    return __builtin_amdgcn_wmma_f32_16x16x32_bf16(false, a, false, b,
                                                   (short)0, c, false, false);
}

// ---------------------------------------------------------------------------
// Kernel 0: f32 -> packed bf16 conversion (bandwidth-bound).
// ---------------------------------------------------------------------------
__global__ __launch_bounds__(256)
void cvt_bf16_kernel(const float* __restrict__ src, unsigned* __restrict__ dst, int nPairs)
{
    const int i = blockIdx.x * 256 + threadIdx.x;
    if (i < nPairs) {
        const float2 f = ((const float2*)src)[i];
        dst[i] = ((unsigned)f2bf(f.y) << 16) | (unsigned)f2bf(f.x);
    }
}

// ---------------------------------------------------------------------------
// Kernel 1: [8192 x 1024] = xbf @ wbf^T (+bias), RoPE on k half.
// One wave = 32x32 C block (2x2 WMMA tiles).  Fragment layouts as loaded here (contiguous K):
//  A(16x32 bf16): lane l: M=l%16; vgpr v: K pair k0+(l/16)*16+2v
//  B(32x16 bf16): lane l: N=l%16; vgpr v: K pair k0+(l/16)*16+2v
//  C(16x16 f32) : lane l: N=l%16; vgpr r: M=r+8*(l/16)
// ---------------------------------------------------------------------------
__device__ __forceinline__ void qk_epilogue(float bv, int mBase, int colB, int half,
                                            const v8f& c,
                                            unsigned short* __restrict__ qbf,
                                            unsigned short* __restrict__ kbf,
                                            unsigned short* __restrict__ kbfT)
{
    if (colB < EMBED) {                                  // q half
#pragma unroll
        for (int r = 0; r < 8; ++r) {
            const int mg = mBase + r + 8 * half;
            qbf[(size_t)mg * EMBED + colB] = f2bf(c[r] + bv);
        }
    } else {                                             // k half: RoPE
        const int col = colB - EMBED;
        const int i   = col >> 1;
        const float theta = __expf(-9.210340371976184f * (2.0f * (float)i) / (float)EMBED);
#pragma unroll
        for (int r = 0; r < 8; ++r) {
            const int mg = mBase + r + 8 * half;
            const int t  = mg & (TLEN - 1);
            const int b  = mg >> 9;
            const float val     = c[r] + bv;
            const float partner = __shfl_xor(val, 1, 32);
            float sn, cs;
            __sincosf((float)(t + 1) * theta, &sn, &cs);
            const float out = (col & 1) ? (cs * val - sn * partner)
                                        : (cs * val + sn * partner);
            kbf [(size_t)mg * EMBED + col]               = f2bf(out);
            kbfT[((size_t)(b * EMBED) + col) * TLEN + t] = f2bf(out);
        }
    }
}

__global__ __launch_bounds__(256)
void qk_gemm_kernel(const unsigned short* __restrict__ xbf,
                    const unsigned short* __restrict__ wbf,
                    const float* __restrict__ bias,
                    unsigned short* __restrict__ qbf,
                    unsigned short* __restrict__ kbf,
                    unsigned short* __restrict__ kbfT)
{
    const int wave = blockIdx.x * 8 + (threadIdx.x >> 5);
    const int lane = threadIdx.x & 31;
    const int nBlocks = (2 * EMBED) / 32;       // 32
    const int mb = wave / nBlocks;              // 0..255
    const int nb = wave % nBlocks;              // 0..31
    const int n = lane & 15, half = lane >> 4;
    const int rowA0 = mb * 32 + n;
    const int colB0 = nb * 32 + n;

    v8f c00 = {}, c01 = {}, c10 = {}, c11 = {};
    for (int k0 = 0; k0 < EMBED; k0 += 32) {
        v8u a0, a1, b0, b1;
#pragma unroll
        for (int v = 0; v < 8; ++v) {
            const int ka = k0 + half * 16 + v * 2;
            a0[v] = *(const unsigned*)(xbf + (size_t)rowA0 * EMBED + ka);
            a1[v] = *(const unsigned*)(xbf + (size_t)(rowA0 + 16) * EMBED + ka);
            const int kb = k0 + half * 16 + v * 2;
            b0[v] = *(const unsigned*)(wbf + (size_t)colB0 * EMBED + kb);
            b1[v] = *(const unsigned*)(wbf + (size_t)(colB0 + 16) * EMBED + kb);
        }
        c00 = bf16_wmma(a0, b0, c00);
        c01 = bf16_wmma(a0, b1, c01);
        c10 = bf16_wmma(a1, b0, c10);
        c11 = bf16_wmma(a1, b1, c11);
    }
    const float bv0 = bias[colB0], bv1 = bias[colB0 + 16];
    qk_epilogue(bv0, mb * 32,      colB0,      half, c00, qbf, kbf, kbfT);
    qk_epilogue(bv1, mb * 32,      colB0 + 16, half, c01, qbf, kbf, kbfT);
    qk_epilogue(bv0, mb * 32 + 16, colB0,      half, c10, qbf, kbf, kbfT);
    qk_epilogue(bv1, mb * 32 + 16, colB0 + 16, half, c11, qbf, kbf, kbfT);
}

// ---------------------------------------------------------------------------
// Kernel 2: causal scores + softmax stats.  One wave = (batch, 16 q rows).
// The wave's 16x512 bf16 q strip is staged to LDS once with async-to-LDS
// (ASYNCcnt), then reused across all key tiles and both passes.
// Pass 0 = row maxima; pass 1 = store unnormalized exp as bf16 + row sums.
// ---------------------------------------------------------------------------
__global__ __launch_bounds__(64)
void attn_softmax_kernel(const unsigned short* __restrict__ qbf,
                         const unsigned short* __restrict__ kbf,
                         unsigned short* __restrict__ pbuf,
                         float* __restrict__ rowsum)
{
    __shared__ unsigned short qs[2][16 * EMBED];        // 32 KiB (2 waves/block)
    const int wslot = threadIdx.x >> 5;
    const int wave  = blockIdx.x * 2 + wslot;
    const int lane  = threadIdx.x & 31;
    const int b     = wave >> 5;
    const int tTile = wave & 31;
    const int n     = lane & 15;
    const int half  = lane >> 4;
    const float scale = 0.044194173824159216f;          // 1/sqrt(512)
    const size_t base = (size_t)b * TLEN * EMBED;
    const int sEnd  = tTile | 1;                        // pad to 32-wide K for K3

    // ---- async stage: 16 rows x 512 bf16 = 16 KiB per wave ----
    {
        const unsigned short* gsrc = qbf + base + (size_t)(tTile * 16) * EMBED;
        const unsigned ldsBase = (unsigned)(uintptr_t)&qs[wslot][0];
#pragma unroll 4
        for (int j = 0; j < 32; ++j) {
            const unsigned off = (unsigned)((j * 32 + lane) * 16);
            const unsigned long long ga = (unsigned long long)(uintptr_t)gsrc + off;
            const unsigned la = ldsBase + off;
            asm volatile("global_load_async_to_lds_b128 %0, %1, off"
                         :: "v"(la), "v"(ga) : "memory");
        }
        asm volatile("s_wait_asynccnt 0x0" ::: "memory");
    }

    float rmax[8];
#pragma unroll
    for (int r = 0; r < 8; ++r) rmax[r] = -3.0e38f;
    float rsum[8];

    for (int pass = 0; pass < 2; ++pass) {
#pragma unroll
        for (int r = 0; r < 8; ++r) rsum[r] = 0.0f;
        for (int sTile = 0; sTile <= sEnd; ++sTile) {
            v8f c = {};
            for (int k0 = 0; k0 < EMBED; k0 += 32) {
                v8u au, bu;
#pragma unroll
                for (int v = 0; v < 8; ++v) {
                    const int ka = k0 + half * 16 + v * 2;
                    au[v] = *(const unsigned*)(&qs[wslot][n * EMBED + ka]);   // ds_load
                    const int kb   = k0 + half * 16 + v * 2;
                    const int srow = sTile * 16 + n;
                    bu[v] = *(const unsigned*)(kbf + base + (size_t)srow * EMBED + kb);
                }
                c = bf16_wmma(au, bu, c);
            }
#pragma unroll
            for (int r = 0; r < 8; ++r) {
                const int t = tTile * 16 + r + 8 * half;
                const int s = sTile * 16 + n;
                const float xv   = c[r] * scale;
                const bool valid = (s <= t);
                if (pass == 0) {
                    float mv = valid ? xv : -3.0e38f;
#pragma unroll
                    for (int m = 1; m < 16; m <<= 1) mv = fmaxf(mv, __shfl_xor(mv, m, 32));
                    rmax[r] = fmaxf(rmax[r], mv);
                } else {
                    const float e = valid ? __expf(xv - rmax[r]) : 0.0f;
                    float es = e;
#pragma unroll
                    for (int m = 1; m < 16; m <<= 1) es += __shfl_xor(es, m, 32);
                    rsum[r] += es;
                    pbuf[((size_t)(b * TLEN) + t) * TLEN + s] = f2bf(e); // zeros where masked
                }
            }
        }
        if (pass == 1 && n == 0) {
#pragma unroll
            for (int r = 0; r < 8; ++r) {
                const int t = tTile * 16 + r + 8 * half;
                rowsum[b * TLEN + t] = rsum[r];
            }
        }
    }
}

// ---------------------------------------------------------------------------
// Kernel 3: y = (expP @ krot) / rowsum.  One wave = 32x32 C block.
// B fragments come from the transposed krot copy -> contiguous dword loads.
// ---------------------------------------------------------------------------
__global__ __launch_bounds__(256)
void pv_gemm_kernel(const unsigned short* __restrict__ pbuf,
                    const unsigned short* __restrict__ kbfT,
                    const float* __restrict__ rowsum,
                    float* __restrict__ out)
{
    const int wave  = blockIdx.x * 8 + (threadIdx.x >> 5);
    const int lane  = threadIdx.x & 31;
    const int dB    = EMBED / 32;               // 16
    const int tB    = TLEN / 32;                // 16
    const int b     = wave / (dB * tB);
    const int rem   = wave % (dB * tB);
    const int tb    = rem / dB;
    const int db    = rem % dB;
    const int n     = lane & 15;
    const int half  = lane >> 4;
    const size_t pb   = (size_t)b * TLEN * TLEN;
    const size_t ktb  = (size_t)b * EMBED * TLEN;
    const int trow0 = tb * 32 + n;
    const int dcol0 = db * 32 + n;
    const int kmax  = (tb + 1) * 32;            // causal K bound (P padded w/ zeros)

    v8f c00 = {}, c01 = {}, c10 = {}, c11 = {};
    for (int k0 = 0; k0 < kmax; k0 += 32) {
        v8u a0, a1, b0, b1;
#pragma unroll
        for (int v = 0; v < 8; ++v) {
            const int ka = k0 + half * 16 + v * 2;
            a0[v] = *(const unsigned*)(pbuf + pb + (size_t)trow0 * TLEN + ka);
            a1[v] = *(const unsigned*)(pbuf + pb + (size_t)(trow0 + 16) * TLEN + ka);
            const int s = k0 + half * 16 + v * 2;
            b0[v] = *(const unsigned*)(kbfT + ktb + (size_t)dcol0 * TLEN + s);
            b1[v] = *(const unsigned*)(kbfT + ktb + (size_t)(dcol0 + 16) * TLEN + s);
        }
        c00 = bf16_wmma(a0, b0, c00);
        c01 = bf16_wmma(a0, b1, c01);
        c10 = bf16_wmma(a1, b0, c10);
        c11 = bf16_wmma(a1, b1, c11);
    }
#pragma unroll
    for (int r = 0; r < 8; ++r) {
        const int t0 = tb * 32 + r + 8 * half;
        const int t1 = t0 + 16;
        const float i0 = 1.0f / rowsum[b * TLEN + t0];
        const float i1 = 1.0f / rowsum[b * TLEN + t1];
        out[((size_t)(b * TLEN) + t0) * EMBED + dcol0]      = c00[r] * i0;
        out[((size_t)(b * TLEN) + t0) * EMBED + dcol0 + 16] = c01[r] * i0;
        out[((size_t)(b * TLEN) + t1) * EMBED + dcol0]      = c10[r] * i1;
        out[((size_t)(b * TLEN) + t1) * EMBED + dcol0 + 16] = c11[r] * i1;
    }
}

// ---------------------------------------------------------------------------
extern "C" void kernel_launch(void* const* d_in, const int* in_sizes, int n_in,
                              void* d_out, int out_size, void* d_ws, size_t ws_size,
                              hipStream_t stream)
{
    (void)in_sizes; (void)n_in; (void)out_size; (void)ws_size;
    const float* x    = (const float*)d_in[0];   // [16,512,512]
    const float* W    = (const float*)d_in[1];   // [1536,512]
    const float* bias = (const float*)d_in[2];   // [1536]
    float* out        = (float*)d_out;           // [16,512,512]

    char* ws = (char*)d_ws;
    const size_t MB = 1024u * 1024u;
    unsigned short* qbf  = (unsigned short*)(ws);             //  8 MiB bf16 q
    unsigned short* kbf  = (unsigned short*)(ws +  8 * MB);   //  8 MiB bf16 krot
    unsigned short* pbuf = (unsigned short*)(ws + 16 * MB);   //  8 MiB bf16 exp(P)
    unsigned short* kbfT = (unsigned short*)(ws + 24 * MB);   //  8 MiB bf16 krot^T
    unsigned*       xbf  = (unsigned*)      (ws + 32 * MB);   //  8 MiB bf16 x (packed)
    unsigned*       wbf  = (unsigned*)      (ws + 40 * MB);   //  1 MiB bf16 W[0:1024]
    float*          rsum = (float*)         (ws + 41 * MB);   // 32 KiB row sums

    const int xPairs = BATCH * TLEN * EMBED / 2;              // 2,097,152
    const int wPairs = 2 * EMBED * EMBED / 2;                 //   262,144
    cvt_bf16_kernel<<<dim3((xPairs + 255) / 256), dim3(256), 0, stream>>>(x, xbf, xPairs);
    cvt_bf16_kernel<<<dim3((wPairs + 255) / 256), dim3(256), 0, stream>>>(W, wbf, wPairs);

    // K1: 256*32 waves, 8 waves/block
    qk_gemm_kernel<<<dim3(1024), dim3(256), 0, stream>>>(
        (const unsigned short*)xbf, (const unsigned short*)wbf, bias, qbf, kbf, kbfT);
    // K2: 512 waves, 2 waves/block (32 KiB LDS each)
    attn_softmax_kernel<<<dim3(256), dim3(64), 0, stream>>>(qbf, kbf, pbuf, rsum);
    // K3: 16*16*16 waves, 8 waves/block
    pv_gemm_kernel<<<dim3(512), dim3(256), 0, stream>>>(pbuf, kbfT, rsum, out);
}
